// OriginalImplementationFastAPLoss_27814208209511
// MI455X (gfx1250) — matrix operand order repaired and checked by reading, not compiled, not run on hardware
//
#include <hip/hip_runtime.h>

// FastAP loss, fused for MI455X (gfx1250):
//   G = X Xt via v_wmma_f32_16x16x32_f16 (f32 accumulate), histogram fused in
//   LDS (2-bin triangular-pulse scatter), tiny per-row AP reduction.
//
// ws layout:
//   [0, 4 MB)            : Xh  f16 [4096][512]
//   [4 MB, +360448)      : Hg  f32 [4096][22]   (entry = bin*2 + isNeg)
//   [next 8 bytes]       : sums[2] = { sum(fastap), n_valid }

#define N_ROWS   4096
#define D_DIM    512
#define NBINS    10
#define LBINS    11            // NBINS + 1 bin centers
#define TILE     128
#define HSTRIDE  24            // 22 counters per row, padded

typedef __attribute__((ext_vector_type(16))) _Float16 v16h;
typedef __attribute__((ext_vector_type(8)))  _Float16 v8h;
typedef __attribute__((ext_vector_type(4)))  _Float16 v4h;
typedef __attribute__((ext_vector_type(8)))  float    v8f;

static constexpr size_t XH_BYTES  = (size_t)N_ROWS * D_DIM * 2;        // 4 MB
static constexpr size_t HG_ELEMS  = (size_t)N_ROWS * 22;               // 90112
static constexpr size_t HG_OFF    = XH_BYTES;
static constexpr size_t SUM_OFF   = HG_OFF + HG_ELEMS * sizeof(float);

// ---------------------------------------------------------------- zero ws ---
__global__ void fastap_zero(float* __restrict__ p, int n) {
    int i = blockIdx.x * 256 + threadIdx.x;
    if (i < n) p[i] = 0.0f;
}

// ------------------------------------------------------------ f32 -> f16 ----
__global__ void fastap_cvt(const float* __restrict__ x, _Float16* __restrict__ xh, int n4) {
    int i = blockIdx.x * 256 + threadIdx.x;
    if (i < n4) {
        float4 v = reinterpret_cast<const float4*>(x)[i];
        v4h h = { (_Float16)v.x, (_Float16)v.y, (_Float16)v.z, (_Float16)v.w };
        reinterpret_cast<v4h*>(xh)[i] = h;
    }
}

// ------------------------------- fused WMMA gram + soft-histogram scatter ---
// Block = 256 threads = 8 waves. Block tile: 128 rows x 128 cols of G.
// Wave w owns rows [tileRow + 16w, +16) and sweeps all 8 column sub-tiles.
__global__ __launch_bounds__(256) void fastap_main(
        const _Float16* __restrict__ Xh,
        const int*      __restrict__ labels,
        float*          __restrict__ Hg)
{
    __shared__ float hloc[TILE * HSTRIDE];         // 12 KB

    const int tid = threadIdx.x;
    for (int i = tid; i < TILE * HSTRIDE; i += 256) hloc[i] = 0.0f;
    __syncthreads();

    const int wave = tid >> 5;
    const int lane = tid & 31;
    const int lo   = lane & 15;
    const int half = lane >> 4;

    const int tileRow = blockIdx.y * TILE;
    const int tileCol = blockIdx.x * TILE;
    const int rowBase = tileRow + wave * 16;

    v8f acc[8];
    #pragma unroll
    for (int c = 0; c < 8; ++c) acc[c] = (v8f){0.f,0.f,0.f,0.f,0.f,0.f,0.f,0.f};

    // A-fragment source row for this lane (M = lo), per ISA 16-bit A layout.
    const _Float16* arow = Xh + (size_t)(rowBase + lo) * D_DIM;

    for (int k0 = 0; k0 < D_DIM; k0 += 32) {
        // A 16x32 f16 frag: elems 0..7  = K[k0 + 16*half .. +8)
        //                   elems 8..15 = K[k0 + 16*half + 8 .. +8)
        v8h a0 = *reinterpret_cast<const v8h*>(arow + k0 + half * 16);
        v8h a1 = *reinterpret_cast<const v8h*>(arow + k0 + half * 16 + 8);
        v16h afrag;
        #pragma unroll
        for (int t = 0; t < 8; ++t) { afrag[t] = a0[t]; afrag[t + 8] = a1[t]; }

        #pragma unroll
        for (int c = 0; c < 8; ++c) {
            // B = Xt: frag for lane (N = lo) is 16 contiguous halfs of row
            // (tileCol + 16c + lo), K = k0 + 16*half + j  -> plain b128 loads.
            const _Float16* brow = Xh + (size_t)(tileCol + c * 16 + lo) * D_DIM;
            v16h bfrag = *reinterpret_cast<const v16h*>(brow + k0 + half * 16);
            acc[c] = __builtin_amdgcn_wmma_f32_16x16x32_f16(
                         false, afrag, false, bfrag,
                         (short)0, acc[c], false, false);
        }
    }

    // Epilogue: C/D layout -> lane holds (row = r + 8*half, col = lo) of tile.
    int li[8];
    #pragma unroll
    for (int r = 0; r < 8; ++r)
        li[r] = labels[tileRow + wave * 16 + r + half * 8];

    #pragma unroll
    for (int c = 0; c < 8; ++c) {
        const int j  = tileCol + c * 16 + lo;
        const int lj = labels[j];
        #pragma unroll
        for (int r = 0; r < 8; ++r) {
            const int ilocal = wave * 16 + r + half * 8;
            const int i      = tileRow + ilocal;
            float d2 = 2.0f - 2.0f * acc[c][r];
            float t  = d2 * ((float)NBINS / 4.0f);          // / delta
            t = fminf(fmaxf(t, 0.0f), (float)NBINS);
            int b0 = (int)t; if (b0 > NBINS - 1) b0 = NBINS - 1;
            float frac = t - (float)b0;
            if (i != j) {
                const int neg = (li[r] != lj) ? 1 : 0;
                float* base = &hloc[ilocal * HSTRIDE + neg];
                atomicAdd(base + 2 * b0,       1.0f - frac);
                atomicAdd(base + 2 * (b0 + 1), frac);
            }
        }
    }
    __syncthreads();

    // Flush per-WG histogram: one global atomic per (row, counter).
    for (int idx = tid; idx < TILE * 22; idx += 256) {
        const int row = idx / 22, rem = idx % 22;
        const float v = hloc[row * HSTRIDE + rem];
        if (v != 0.0f)
            atomicAdd(&Hg[(size_t)(tileRow + row) * 22 + rem], v);
    }
}

// ------------------------------------------- per-row cumsum / AP / reduce ---
__global__ __launch_bounds__(256) void fastap_reduce(
        const float* __restrict__ Hg, float* __restrict__ sums)
{
    const int i = blockIdx.x * 256 + threadIdx.x;
    float fa = 0.0f, valid = 0.0f;
    if (i < N_ROWS) {
        const float* h = Hg + (size_t)i * 22;
        float Hp = 0.0f, Ht = 0.0f, rowap = 0.0f, npos = 0.0f;
        #pragma unroll
        for (int l = 0; l < LBINS; ++l) {
            const float hp = h[2 * l], hn = h[2 * l + 1];
            Hp += hp; Ht += hp + hn;
            if (Ht > 0.0f) rowap += hp * Hp / Ht;
            npos += hp;                 // pulses form a partition of unity
        }
        if (npos > 0.0f) { fa = rowap / npos; valid = 1.0f; }
    }
    __shared__ float s0[256], s1[256];
    s0[threadIdx.x] = fa; s1[threadIdx.x] = valid;
    __syncthreads();
    for (int ofs = 128; ofs > 0; ofs >>= 1) {
        if (threadIdx.x < ofs) {
            s0[threadIdx.x] += s0[threadIdx.x + ofs];
            s1[threadIdx.x] += s1[threadIdx.x + ofs];
        }
        __syncthreads();
    }
    if (threadIdx.x == 0) {
        atomicAdd(&sums[0], s0[0]);
        atomicAdd(&sums[1], s1[0]);
    }
}

__global__ void fastap_final(const float* __restrict__ sums, float* __restrict__ out) {
    if (threadIdx.x == 0 && blockIdx.x == 0)
        out[0] = 1.0f - sums[0] / sums[1];
}

// ---------------------------------------------------------------------------
extern "C" void kernel_launch(void* const* d_in, const int* in_sizes, int n_in,
                              void* d_out, int out_size, void* d_ws, size_t ws_size,
                              hipStream_t stream) {
    const float* batch  = (const float*)d_in[0];
    const int*   labels = (const int*)d_in[1];
    float*       out    = (float*)d_out;

    char*     ws   = (char*)d_ws;
    _Float16* Xh   = (_Float16*)ws;
    float*    Hg   = (float*)(ws + HG_OFF);
    float*    sums = (float*)(ws + SUM_OFF);

    // 1) zero histogram + scalar accumulators (contiguous: Hg then sums)
    {
        const int n = (int)HG_ELEMS + 2;
        fastap_zero<<<(n + 255) / 256, 256, 0, stream>>>(Hg, n);
    }
    // 2) f32 -> f16 copy of X (4 elements per thread)
    {
        const int n4 = N_ROWS * D_DIM / 4;
        fastap_cvt<<<(n4 + 255) / 256, 256, 0, stream>>>(batch, Xh, n4);
    }
    // 3) fused WMMA gram + soft-histogram
    {
        dim3 grid(N_ROWS / TILE, N_ROWS / TILE);   // 32 x 32
        fastap_main<<<grid, 256, 0, stream>>>(Xh, labels, Hg);
    }
    // 4) per-row AP + global reduction
    fastap_reduce<<<N_ROWS / 256, 256, 0, stream>>>(Hg, sums);
    // 5) final scalar
    fastap_final<<<1, 32, 0, stream>>>(sums, out);
}
